// CaptioningRNN_84885733638384
// MI455X (gfx1250) — matrix operand order repaired and checked by reading, not compiled, not run on hardware
//
#include <hip/hip_runtime.h>
#include <hip/hip_bf16.h>
#include <math.h>

typedef __attribute__((ext_vector_type(8)))  _Float16 v8h;
typedef __attribute__((ext_vector_type(16))) _Float16 v16h;
typedef __attribute__((ext_vector_type(8)))  float    v8f;

#define N_BATCH 64
#define T_SEQ   256
#define TM1     255
#define VOCAB   10000
#define DFEAT   1280
#define WDIM    256
#define HDIM    512
#define M_ROWS  (N_BATCH * TM1)   // 16320
#define M_TILES (M_ROWS / 16)     // 1020
#define STRIP   64                // scores M-strip rows per block
#define NSTRIPS (M_ROWS / STRIP)  // 255
#define VCHUNK  256
#define NCHUNKS 40                // 40*256 = 10240 >= 10000
#define NEGBIG  (-3.0e38f)

// ---------------- WMMA tile helpers (wave32, 16x16x32 f16 -> f32) ----------------

__device__ __forceinline__ v16h hcat(v8h lo, v8h hi) {
  v16h r;
#pragma unroll
  for (int i = 0; i < 8; ++i) { r[i] = lo[i]; r[i + 8] = hi[i]; }
  return r;
}

// A matrix 16x32 f16, row-major source with leading dim lda.
// Lane l: row = l&15; elements 0..15 -> K = kbase + 16*(l>>4) + 0..15 (contiguous)
__device__ __forceinline__ v16h load_a_rowmajor(const _Float16* A, int lda, int kbase) {
  int lane = threadIdx.x & 31;
  int row  = lane & 15;
  int k0   = kbase + ((lane >> 4) << 4);
  const _Float16* p = A + (size_t)row * lda + k0;
  v8h lo = *(const v8h*)(p);
  v8h hi = *(const v8h*)(p + 8);
  return hcat(lo, hi);
}

// B matrix 32x16 (KxN) from transposed storage BT[N][K] (leading dim ldb=K).
// Lane l: col = colbase + (l&15); K = kbase + 16*(l>>4) + 0..15 (contiguous).
__device__ __forceinline__ v16h load_b_trans(const _Float16* BT, int ldb, int kbase, int colbase) {
  int lane = threadIdx.x & 31;
  int col  = colbase + (lane & 15);
  int k0   = kbase + ((lane >> 4) << 4);
  return *(const v16h*)(BT + (size_t)col * ldb + k0);
}

__device__ __forceinline__ v8f wmma_f16(v16h a, v16h b, v8f c) {
  return __builtin_amdgcn_wmma_f32_16x16x32_f16(false, a, false, b, (short)0, c, false, false);
}

// C layout: element i of lane l -> row = i + 8*(l>=16), col = l&15

// online-softmax partial merge: (m,s) <- combine((m,s),(m2,s2)); s==0 means empty
__device__ __forceinline__ void combine_ms(float& m, float& s, float m2, float s2) {
  if (s2 <= 0.0f) return;
  if (s  <= 0.0f) { m = m2; s = s2; return; }
  float nm = fmaxf(m, m2);
  s = s * __expf(m - nm) + s2 * __expf(m2 - nm);
  m = nm;
}

// ---------------- CDNA5 async copy (global -> LDS, ASYNCcnt-tracked) -------------

__device__ __forceinline__ unsigned lds_off_u32(void* p) {
  // generic -> LDS addrspace; hardware adds the wave's LDS_BASE to VGPR[VDST]
  return (unsigned)(size_t)(__attribute__((address_space(3))) void*)p;
}

__device__ __forceinline__ void async_copy_b128(unsigned lds_off, const void* gaddr) {
  asm volatile("global_load_async_to_lds_b128 %0, %1, off"
               :: "v"(lds_off), "v"((unsigned long long)(size_t)gaddr)
               : "memory");
}

__device__ __forceinline__ void wait_async_all() {
  asm volatile("s_wait_asynccnt 0x0" ::: "memory");
}

// ---------------- conversion kernels ----------------

__global__ void cvt_f32_f16(const float* __restrict__ src, _Float16* __restrict__ dst, int n) {
  int i = blockIdx.x * blockDim.x + threadIdx.x;
  if (i < n) dst[i] = (_Float16)src[i];
}

// src: K x N row-major  ->  dst: N x K row-major (transposed)
__global__ void cvtT_f32_f16(const float* __restrict__ src, _Float16* __restrict__ dst, int K, int N) {
  int i = blockIdx.x * blockDim.x + threadIdx.x;
  if (i < K * N) {
    int k = i / N, n = i % N;
    dst[(size_t)n * K + k] = (_Float16)src[i];
  }
}

__global__ void zero_out(float* out) { if (threadIdx.x == 0) out[0] = 0.0f; }

// ---------------- h0 = feat @ W_proj + b_proj  (64x512, K=1280) ----------------

__global__ void gemm_h0(const _Float16* __restrict__ feat16, const _Float16* __restrict__ WprojT,
                        const float* __restrict__ b_proj, float* __restrict__ H0) {
  int wid  = (blockIdx.x * blockDim.x + threadIdx.x) >> 5;  // 0..127
  int lane = threadIdx.x & 31;
  int mt = wid & 3;        // 4 M-tiles
  int nt = wid >> 2;       // 32 N-tiles
  const _Float16* A = feat16 + (size_t)mt * 16 * DFEAT;
  v8f acc = {};
#pragma unroll 4
  for (int k = 0; k < DFEAT; k += 32) {
    v16h a = load_a_rowmajor(A, DFEAT, k);
    v16h b = load_b_trans(WprojT, DFEAT, k, nt * 16);
    acc = wmma_f16(a, b, acc);
  }
  int col = nt * 16 + (lane & 15);
  float bias = b_proj[col];
#pragma unroll
  for (int i = 0; i < 8; ++i) {
    int row = mt * 16 + i + ((lane >> 4) << 3);
    H0[(size_t)row * HDIM + col] = acc[i] + bias;
  }
}

// ---------------- xW = W_embed[cap_in] @ Wx + b  (16320x512, K=256) ----------------

__global__ void gemm_xw(const _Float16* __restrict__ Wemb16, const _Float16* __restrict__ WxT,
                        const float* __restrict__ bvec, const int* __restrict__ captions,
                        _Float16* __restrict__ xW16) {
  int wid  = (blockIdx.x << 3) + (threadIdx.x >> 5);  // 8 waves/block, 32640 tiles
  int lane = threadIdx.x & 31;
  int mt = wid >> 5;       // 0..1019
  int nt = wid & 31;       // 0..31
  // per-lane gathered A row: flat row m = n*255 + t ; cap_in[n,t] = captions[n*256+t]
  int m   = mt * 16 + (lane & 15);
  int n   = m / TM1, t = m % TM1;
  const _Float16* arow = Wemb16 + (size_t)captions[n * T_SEQ + t] * WDIM;
  int koffA = (lane >> 4) << 4;
  v8f acc = {};
#pragma unroll
  for (int k = 0; k < WDIM; k += 32) {
    v8h lo = *(const v8h*)(arow + k + koffA);
    v8h hi = *(const v8h*)(arow + k + koffA + 8);
    v16h a = hcat(lo, hi);
    v16h b = load_b_trans(WxT, WDIM, k, nt * 16);
    acc = wmma_f16(a, b, acc);
  }
  int col = nt * 16 + (lane & 15);
  float bias = bvec[col];
#pragma unroll
  for (int i = 0; i < 8; ++i) {
    int row = mt * 16 + i + ((lane >> 4) << 3);
    xW16[(size_t)row * HDIM + col] = (_Float16)(acc[i] + bias);
  }
}

// ---------------- RNN scan: h = tanh(xW_t + h @ Wh), per-sample independent ------
// 4 blocks x 1024 threads (32 waves). Block owns 16 sample rows; wave owns one N-tile.
// Dynamic LDS: double-buffered h tile, 2 * 16 * 512 f16 = 32 KB.

__global__ void rnn_scan(const float* __restrict__ H0, const _Float16* __restrict__ xW16,
                         const _Float16* __restrict__ WhT, _Float16* __restrict__ hs16) {
  extern __shared__ _Float16 hsh[];   // [2][16][HDIM]
  _Float16* buf0 = hsh;
  _Float16* buf1 = hsh + 16 * HDIM;
  int m0 = blockIdx.x * 16;           // sample rows m0..m0+15

  for (int i = threadIdx.x; i < 16 * HDIM; i += blockDim.x) {
    int r = i >> 9, c = i & (HDIM - 1);
    buf0[i] = (_Float16)H0[(size_t)(m0 + r) * HDIM + c];
  }
  __syncthreads();

  int wave = threadIdx.x >> 5;        // N-tile 0..31
  int lane = threadIdx.x & 31;
  int col  = wave * 16 + (lane & 15);
  int cur  = 0;

  for (int t = 0; t < TM1; ++t) {
    const _Float16* hb = cur ? buf1 : buf0;
    _Float16*       hn = cur ? buf0 : buf1;
    v8f acc = {};
#pragma unroll 4
    for (int k = 0; k < HDIM; k += 32) {
      v16h a = load_a_rowmajor(hb, HDIM, k);          // from LDS
      v16h b = load_b_trans(WhT, HDIM, k, wave * 16); // from L2
      acc = wmma_f16(a, b, acc);
    }
#pragma unroll
    for (int i = 0; i < 8; ++i) {
      int r = i + ((lane >> 4) << 3);                 // row within strip
      int nrow = m0 + r;                              // sample index
      size_t mrow = (size_t)nrow * TM1 + t;           // flat (n,t) row
      float v = acc[i] + (float)xW16[mrow * HDIM + col];
      _Float16 hv = (_Float16)tanhf(v);
      hs16[mrow * HDIM + col] = hv;
      hn[r * HDIM + col] = hv;
    }
    __syncthreads();
    cur ^= 1;
  }
}

// ---------------- scores = hs @ W_out + b_out -> online (max,sumexp) + picked -----
// Grid (255 strips, 40 V-chunks), 512 threads = 16 waves.
// Wave owns one N-tile, computes 4 M-tiles reusing one B register load per k-step.
// A strip (64x512 f16 = 64KB) staged into LDS via CDNA5 async global->LDS copies.

__global__ void scores_lse(const _Float16* __restrict__ hs16, const _Float16* __restrict__ WoutT,
                           const float* __restrict__ b_out, const int* __restrict__ captions,
                           float2* __restrict__ partial, float* __restrict__ picked) {
  extern __shared__ char smem[];
  _Float16* Ash = (_Float16*)smem;                          // STRIP*HDIM f16 = 64KB
  float*    red = (float*)(smem + STRIP * HDIM * 2);        // 16*STRIP*2 f32 = 8KB
  int*      tgt = (int*)(smem + STRIP * HDIM * 2 + 16 * STRIP * 2 * 4);

  int strip = blockIdx.x;        // 0..254
  int chunk = blockIdx.y;        // 0..39
  int m0 = strip * STRIP;

  // --- async stage: 64KB contiguous strip of hs16 -> LDS (16B per lane per issue)
  {
    const char* gsrc = (const char*)(hs16 + (size_t)m0 * HDIM);
    unsigned lbase = lds_off_u32(Ash);
    const int nchunks16 = (STRIP * HDIM * 2) / 16;          // 4096
    for (int c = threadIdx.x; c < nchunks16; c += blockDim.x)
      async_copy_b128(lbase + c * 16, gsrc + (size_t)c * 16);
    wait_async_all();
  }
  if (threadIdx.x < STRIP) {
    int m = m0 + threadIdx.x;
    int n = m / TM1, t = m % TM1;
    tgt[threadIdx.x] = captions[n * T_SEQ + t + 1];         // cap_out
  }
  __syncthreads();

  int wave = threadIdx.x >> 5;   // 0..15 -> N-tile within chunk
  int lane = threadIdx.x & 31;
  int col  = chunk * VCHUNK + wave * 16 + (lane & 15);      // global vocab col
  int colc = col < VOCAB ? col : (VOCAB - 1);               // clamp load, mask later
  bool valid = col < VOCAB;
  int koffB = (lane >> 4) << 4;
  const _Float16* bcol = WoutT + (size_t)colc * HDIM;

  v8f acc[4] = {{}, {}, {}, {}};
#pragma unroll 4
  for (int k = 0; k < HDIM; k += 32) {
    v16h b = *(const v16h*)(bcol + k + koffB);              // one B load ...
#pragma unroll
    for (int mt4 = 0; mt4 < 4; ++mt4) {                     // ... reused by 4 WMMAs
      v16h a = load_a_rowmajor(Ash + (size_t)mt4 * 16 * HDIM, HDIM, k);
      acc[mt4] = wmma_f16(a, b, acc[mt4]);
    }
  }

  float bias = valid ? b_out[col] : 0.0f;
#pragma unroll
  for (int mt4 = 0; mt4 < 4; ++mt4) {
#pragma unroll
    for (int i = 0; i < 8; ++i) {
      int r = mt4 * 16 + i + ((lane >> 4) << 3);            // row within strip
      float v = valid ? (acc[mt4][i] + bias) : NEGBIG;
      if (valid && col == tgt[r]) picked[m0 + r] = v;
      // reduce over the 16 cols of this N-tile living in this half-wave
      float mx = v;
#pragma unroll
      for (int d = 1; d < 16; d <<= 1) mx = fmaxf(mx, __shfl_xor(mx, d, 16));
      float e = (v > -1.0e37f) ? __expf(v - mx) : 0.0f;
#pragma unroll
      for (int d = 1; d < 16; d <<= 1) e += __shfl_xor(e, d, 16);
      if ((lane & 15) == 0) {
        red[(wave * STRIP + r) * 2 + 0] = mx;
        red[(wave * STRIP + r) * 2 + 1] = e;
      }
    }
  }
  __syncthreads();
  if (threadIdx.x < STRIP) {
    int r = threadIdx.x;
    float m = NEGBIG, s = 0.0f;
#pragma unroll
    for (int w = 0; w < 16; ++w)
      combine_ms(m, s, red[(w * STRIP + r) * 2 + 0], red[(w * STRIP + r) * 2 + 1]);
    partial[(size_t)(m0 + r) * NCHUNKS + chunk] = make_float2(m, s);
  }
}

// ---------------- final: merge chunks -> lse, loss = sum mask*(lse - picked)/N ----

__global__ void final_reduce(const float2* __restrict__ partial, const float* __restrict__ picked,
                             const int* __restrict__ captions, float* __restrict__ out) {
  float local = 0.0f;
  for (int m = blockIdx.x * blockDim.x + threadIdx.x; m < M_ROWS; m += gridDim.x * blockDim.x) {
    int n = m / TM1, t = m % TM1;
    int tg = captions[n * T_SEQ + t + 1];
    if (tg != 0) {
      float mx = NEGBIG, s = 0.0f;
      for (int c = 0; c < NCHUNKS; ++c) {
        float2 p = partial[(size_t)m * NCHUNKS + c];
        combine_ms(mx, s, p.x, p.y);
      }
      float lse = mx + __logf(s);
      local += (lse - picked[m]);
    }
  }
  __shared__ float red[256];
  red[threadIdx.x] = local;
  __syncthreads();
  for (int s2 = 128; s2 > 0; s2 >>= 1) {
    if ((int)threadIdx.x < s2) red[threadIdx.x] += red[threadIdx.x + s2];
    __syncthreads();
  }
  if (threadIdx.x == 0) atomicAdd(out, red[0] / (float)N_BATCH);
}

// ---------------- host ----------------

extern "C" void kernel_launch(void* const* d_in, const int* in_sizes, int n_in,
                              void* d_out, int out_size, void* d_ws, size_t ws_size,
                              hipStream_t stream) {
  (void)in_sizes; (void)n_in; (void)out_size; (void)ws_size;
  const float* feat    = (const float*)d_in[0];
  const float* W_proj  = (const float*)d_in[1];
  const float* b_proj  = (const float*)d_in[2];
  const float* W_embed = (const float*)d_in[3];
  const float* Wx      = (const float*)d_in[4];
  const float* Wh      = (const float*)d_in[5];
  const float* bvec    = (const float*)d_in[6];
  const float* W_out   = (const float*)d_in[7];
  const float* b_out   = (const float*)d_in[8];
  const int*   captions= (const int*)d_in[9];

  char* ws = (char*)d_ws;
  size_t off = 0;
  auto carve = [&](size_t bytes) -> char* {
    char* p = ws + off;
    off = (off + bytes + 255) & ~(size_t)255;
    return p;
  };
  _Float16* feat16  = (_Float16*)carve((size_t)N_BATCH * DFEAT * 2);
  _Float16* WprojT  = (_Float16*)carve((size_t)HDIM * DFEAT * 2);
  _Float16* Wemb16  = (_Float16*)carve((size_t)VOCAB * WDIM * 2);
  _Float16* WxT     = (_Float16*)carve((size_t)HDIM * WDIM * 2);
  _Float16* WhT     = (_Float16*)carve((size_t)HDIM * HDIM * 2);
  _Float16* WoutT   = (_Float16*)carve((size_t)VOCAB * HDIM * 2);
  float*    H0      = (float*)carve((size_t)N_BATCH * HDIM * 4);
  _Float16* xW16    = (_Float16*)carve((size_t)M_ROWS * HDIM * 2);
  _Float16* hs16    = (_Float16*)carve((size_t)M_ROWS * HDIM * 2);
  float2*   partial = (float2*)carve((size_t)M_ROWS * NCHUNKS * 8);
  float*    picked  = (float*)carve((size_t)M_ROWS * 4);

  zero_out<<<1, 32, 0, stream>>>((float*)d_out);

  int n;
  n = N_BATCH * DFEAT;
  cvt_f32_f16<<<(n + 255) / 256, 256, 0, stream>>>(feat, feat16, n);
  n = VOCAB * WDIM;
  cvt_f32_f16<<<(n + 255) / 256, 256, 0, stream>>>(W_embed, Wemb16, n);
  n = DFEAT * HDIM;
  cvtT_f32_f16<<<(n + 255) / 256, 256, 0, stream>>>(W_proj, WprojT, DFEAT, HDIM);
  n = WDIM * HDIM;
  cvtT_f32_f16<<<(n + 255) / 256, 256, 0, stream>>>(Wx, WxT, WDIM, HDIM);
  n = HDIM * HDIM;
  cvtT_f32_f16<<<(n + 255) / 256, 256, 0, stream>>>(Wh, WhT, HDIM, HDIM);
  n = HDIM * VOCAB;
  cvtT_f32_f16<<<(n + 255) / 256, 256, 0, stream>>>(W_out, WoutT, HDIM, VOCAB);

  gemm_h0<<<16, 256, 0, stream>>>(feat16, WprojT, b_proj, H0);
  gemm_xw<<<(M_TILES * 32) / 8, 256, 0, stream>>>(Wemb16, WxT, bvec, captions, xW16);

  size_t rnn_lds = (size_t)2 * 16 * HDIM * 2;
  rnn_scan<<<4, 1024, rnn_lds, stream>>>(H0, xW16, WhT, hs16);

  size_t sc_lds = (size_t)STRIP * HDIM * 2 + (size_t)16 * STRIP * 2 * 4 + (size_t)STRIP * 4;
  scores_lse<<<dim3(NSTRIPS, NCHUNKS), 512, sc_lds, stream>>>(hs16, WoutT, b_out, captions,
                                                              partial, picked);

  final_reduce<<<64, 256, 0, stream>>>(partial, picked, captions, (float*)d_out);
}
